// TriangleAttentionStart_89661737271793
// MI455X (gfx1250) — matrix operand order repaired and checked by reading, not compiled, not run on hardware
//
#include <hip/hip_runtime.h>
#include <hip/hip_bf16.h>
#include <math.h>

typedef __attribute__((ext_vector_type(16))) __bf16 v16bf;
typedef __attribute__((ext_vector_type(8)))  float  v8f;
typedef unsigned short u16;
typedef unsigned int   u32;
typedef unsigned long long u64;

#define CZv   128
#define Lv    256
#define NHv   4
#define HDv   32
#define NPIX  65536          // L*L
#define NTOT  8388608        // CZ*NPIX

#define WMMA_BF16(A,B,C) \
  __builtin_amdgcn_wmma_f32_16x16x32_bf16(false,(A),false,(B),(short)0,(C),false,false)

__device__ __forceinline__ u16 f2bf(float f) {
  u32 u = __float_as_uint(f);
  u32 r = (u + 0x7FFFu + ((u >> 16) & 1u)) >> 16;   // round-to-nearest-even
  return (u16)r;
}
__device__ __forceinline__ float bf2f(u16 h) {
  return __uint_as_float(((u32)h) << 16);
}

union AFrag { v16bf v; u16 s[16]; uint4 q[2]; };

// async copy of 16 bytes: global -> LDS, tracked by ASYNCcnt (no VGPR transit)
__device__ __forceinline__ void async_g2l_b128(u32 lds_off, const void* gptr) {
  u64 ga = (u64)(uintptr_t)gptr;
  asm volatile("global_load_async_to_lds_b128 %0, %1, off"
               :: "v"(lds_off), "v"(ga) : "memory");
}
__device__ __forceinline__ void wait_asynccnt0() {
  asm volatile("s_wait_asynccnt 0x0" ::: "memory");
}

// ---------------------------------------------------------------------------
// 1) Global mean/var reduction (two stage, double accumulation)
// ---------------------------------------------------------------------------
__global__ __launch_bounds__(256) void k_reduce(const float4* __restrict__ x,
                                                double* __restrict__ part) {
  int tid = threadIdx.x;
  double s = 0.0, q = 0.0;
  for (long i = blockIdx.x * 256 + tid; i < NTOT / 4; i += (long)gridDim.x * 256) {
    float4 v = x[i];
    s += (double)v.x + (double)v.y + (double)v.z + (double)v.w;
    q += (double)v.x * v.x + (double)v.y * v.y + (double)v.z * v.z + (double)v.w * v.w;
  }
  __shared__ double ss[256], sq[256];
  ss[tid] = s; sq[tid] = q;
  __syncthreads();
  for (int o = 128; o > 0; o >>= 1) {
    if (tid < o) { ss[tid] += ss[tid + o]; sq[tid] += sq[tid + o]; }
    __syncthreads();
  }
  if (tid == 0) { part[blockIdx.x * 2] = ss[0]; part[blockIdx.x * 2 + 1] = sq[0]; }
}

__global__ __launch_bounds__(256) void k_stats(const double* __restrict__ part,
                                               float* __restrict__ stats) {
  int tid = threadIdx.x;
  __shared__ double ss[256], sq[256];
  ss[tid] = part[2 * tid]; sq[tid] = part[2 * tid + 1];
  __syncthreads();
  for (int o = 128; o > 0; o >>= 1) {
    if (tid < o) { ss[tid] += ss[tid + o]; sq[tid] += sq[tid + o]; }
    __syncthreads();
  }
  if (tid == 0) {
    double mu  = ss[0] / (double)NTOT;
    double var = sq[0] / (double)NTOT - mu * mu;
    stats[0] = (float)mu;
    stats[1] = (float)(1.0 / sqrt(var + 1e-5));
  }
}

// ---------------------------------------------------------------------------
// 2) Fused LayerNorm affine + fp32 -> bf16
// ---------------------------------------------------------------------------
__global__ __launch_bounds__(256) void k_norm(const float4* __restrict__ x,
                                              const float4* __restrict__ ga,
                                              const float4* __restrict__ be,
                                              const float* __restrict__ stats,
                                              u16* __restrict__ xn) {
  int i = blockIdx.x * 256 + threadIdx.x;      // NTOT/4 vec4 entries
  float mu = stats[0], rs = stats[1];
  float4 xv = x[i], gv = ga[i], bv = be[i];
  union { uint2 u; u16 s[4]; } o;
  o.s[0] = f2bf((xv.x - mu) * rs * gv.x + bv.x);
  o.s[1] = f2bf((xv.y - mu) * rs * gv.y + bv.y);
  o.s[2] = f2bf((xv.z - mu) * rs * gv.z + bv.z);
  o.s[3] = f2bf((xv.w - mu) * rs * gv.w + bv.w);
  *(uint2*)(xn + (size_t)i * 4) = o.u;
}

// ---------------------------------------------------------------------------
// 3) Pack a 128x128 fp32 weight matrix into WMMA A-fragment order (bf16).
// ---------------------------------------------------------------------------
__global__ void k_packw(const float* __restrict__ W, u16* __restrict__ dst) {
  int tile = blockIdx.x;                  // mt*4 + kt, 32 tiles
  int mt = tile >> 2, kt = tile & 3;
  int lane = threadIdx.x;
  int lm = lane & 15, lh = lane >> 4;
  int m = mt * 16 + lm;
  int kb = kt * 32 + lh * 16;
  u16* o = dst + tile * 512 + lane * 16;
#pragma unroll
  for (int e = 0; e < 16; ++e) {
    int k = kb + e;
    o[e] = f2bf(W[m * 128 + k]);
  }
}

// ---------------------------------------------------------------------------
// 4) Attention bias map: bias[h, p] = sum_c w_b[h,c] * xn[c,p]
// ---------------------------------------------------------------------------
__global__ __launch_bounds__(256) void k_bias(const u16* __restrict__ xn,
                                              const float* __restrict__ wb,
                                              float* __restrict__ bb) {
  int p = blockIdx.x * 256 + threadIdx.x;
  float a0 = 0.f, a1 = 0.f, a2 = 0.f, a3 = 0.f;
  for (int c = 0; c < 128; ++c) {
    float xv = bf2f(xn[(size_t)c * NPIX + p]);
    a0 = fmaf(wb[c],       xv, a0);
    a1 = fmaf(wb[128 + c], xv, a1);
    a2 = fmaf(wb[256 + c], xv, a2);
    a3 = fmaf(wb[384 + c], xv, a3);
  }
  bb[p] = a0; bb[NPIX + p] = a1; bb[2 * NPIX + p] = a2; bb[3 * NPIX + p] = a3;
}

// ---------------------------------------------------------------------------
// 5) WMMA GEMM: Y[128 x 65536] = Wp(128x128, prepacked bf16) * X(128x65536 bf16)
//    EPI 0: store bf16; EPI 1: sigmoid(y + bvec[m]) -> bf16; EPI 2: y+bvec[m] -> f32
// ---------------------------------------------------------------------------
template <int EPI>
__global__ __launch_bounds__(256) void k_gemm(const u16* __restrict__ Wp,
                                              const u16* __restrict__ X,
                                              void* __restrict__ Yv,
                                              const float* __restrict__ bvec) {
  __shared__ __align__(32) u16 XsT[64][144];   // [n_local][k], row=288B (32B-aligned frags)
  const int n0 = blockIdx.x * 64;
  const int tid = threadIdx.x;

  // transpose-stage X tile (128K x 64N) into LDS, vectorized b128 global loads
  for (int idx = tid; idx < 128 * 8; idx += 256) {
    int k = idx >> 3, grp = idx & 7;
    __builtin_prefetch(X + (size_t)k * NPIX + n0 + 64, 0, 1);
    union { uint4 q; u16 s[8]; } u;
    u.q = *(const uint4*)(X + (size_t)k * NPIX + n0 + grp * 8);
#pragma unroll
    for (int j = 0; j < 8; ++j) XsT[grp * 8 + j][k] = u.s[j];
  }
  __syncthreads();

  const int w = tid >> 5, lane = tid & 31;
  const int lm = lane & 15, lh = lane >> 4;
  const int ntile = w >> 1, mhalf = w & 1;
  const int nl = ntile * 16 + lm;

  v8f acc0 = {0,0,0,0,0,0,0,0}, acc1 = acc0, acc2 = acc0, acc3 = acc0;
#pragma unroll
  for (int kt = 0; kt < 4; ++kt) {
    const v16bf bfrag = *(const v16bf*)&XsT[nl][kt * 32 + lh * 16];
    const u16* wb = Wp + (size_t)(mhalf * 4 * 4 + kt) * 512 + lane * 16;
    v16bf a0 = *(const v16bf*)(wb + 0 * 4 * 512);
    v16bf a1 = *(const v16bf*)(wb + 1 * 4 * 512);
    v16bf a2 = *(const v16bf*)(wb + 2 * 4 * 512);
    v16bf a3 = *(const v16bf*)(wb + 3 * 4 * 512);
    acc0 = WMMA_BF16(a0, bfrag, acc0);
    acc1 = WMMA_BF16(a1, bfrag, acc1);
    acc2 = WMMA_BF16(a2, bfrag, acc2);
    acc3 = WMMA_BF16(a3, bfrag, acc3);
  }

  const int n = n0 + ntile * 16 + lm;
  v8f accs[4] = {acc0, acc1, acc2, acc3};
#pragma unroll
  for (int t = 0; t < 4; ++t) {
#pragma unroll
    for (int r = 0; r < 8; ++r) {
      int m = (mhalf * 4 + t) * 16 + r + 8 * lh;
      float v = accs[t][r];
      size_t oidx = (size_t)m * NPIX + n;
      if constexpr (EPI == 1) {
        v = 1.0f / (1.0f + expf(-(v + bvec[m])));
        ((u16*)Yv)[oidx] = f2bf(v);
      } else if constexpr (EPI == 2) {
        ((float*)Yv)[oidx] = v + bvec[m];
      } else {
        ((u16*)Yv)[oidx] = f2bf(v);
      }
    }
  }
}

// ---------------------------------------------------------------------------
// 6) Fused flash-style triangle attention per (h, i, j-block of 128).
//    V tile staged with GLOBAL_LOAD_ASYNC_TO_LDS_B128 (ASYNCcnt path, no VGPR
//    transit); K tile lane-transposed through VGPRs. One WMMA per 16x16 score
//    tile (K=c=32), online softmax in registers, probs recycled through LDS as
//    bf16 A-fragments for the A*V^T WMMAs. Gating + (c*H+h) repack fused.
// ---------------------------------------------------------------------------
__global__ __launch_bounds__(256) void k_attn(const u16* __restrict__ qb,
                                              const u16* __restrict__ kbuf,
                                              const u16* __restrict__ vbuf,
                                              const u16* __restrict__ gbuf,
                                              const float* __restrict__ biasb,
                                              u16* __restrict__ ob) {
  const int jblk = blockIdx.x, ii = blockIdx.y, h = blockIdx.z;
  __shared__ __align__(32) u16 Ks[256][32];     // [k][c]
  __shared__ __align__(32) u16 Vs[32][256];     // [c][k]
  __shared__ float Stile[8][16][33];            // per-wave raw score strip
  __shared__ __align__(32) u16 Abf[8][16][32];  // per-wave softmaxed probs (bf16)
  __shared__ float rowscale[8][16];
  __shared__ float rowsum[8][16];

  const int tid = threadIdx.x;
  const size_t base = (size_t)(h * HDv) * NPIX + (size_t)ii * Lv;

  // stage K (transposed to [k][c]) via VGPRs; V ([c][k], pure copy) via async DMA
  for (int idx = tid; idx < 32 * 32; idx += 256) {
    int c = idx >> 5, kg = idx & 31;
    async_g2l_b128((u32)(uintptr_t)&Vs[c][kg * 8],
                   vbuf + base + (size_t)c * NPIX + kg * 8);
    union { uint4 q; u16 s[8]; } uk;
    uk.q = *(const uint4*)(kbuf + base + (size_t)c * NPIX + kg * 8);
#pragma unroll
    for (int j = 0; j < 8; ++j) Ks[kg * 8 + j][c] = uk.s[j];
  }
  wait_asynccnt0();
  __syncthreads();

  const int w = tid >> 5, lane = tid & 31;
  const int lm = lane & 15, lh = lane >> 4;
  const int j0 = jblk * 128 + w * 16;

  // Q A-fragment (rows j0..j0+15, K=c=0..31), loaded once, reused all k-tiles
  AFrag qf;
  {
    const int j = j0 + lm;
    const int cb = lh * 16;
#pragma unroll
    for (int e = 0; e < 16; ++e) {
      int c = cb + e;
      qf.s[e] = qb[base + (size_t)c * NPIX + j];
    }
  }

  const float scale = 0.17677669529663687f;   // 1/sqrt(32)
  const float* brow = biasb + (size_t)h * NPIX + (size_t)(j0 + lm) * Lv;
  float m_run = -3.0e38f, l_run = 0.0f;
  v8f accA = {0,0,0,0,0,0,0,0}, accB = accA;

  for (int kc = 0; kc < 8; ++kc) {            // 32 key columns per iteration
    const int kcol0 = kc * 32;
    v8f s0 = {0,0,0,0,0,0,0,0}, s1 = s0;
    {
      const v16bf b0 = *(const v16bf*)&Ks[kcol0 + lm][lh * 16];
      const v16bf b1 = *(const v16bf*)&Ks[kcol0 + 16 + lm][lh * 16];
      s0 = WMMA_BF16(qf.v, b0, s0);
      s1 = WMMA_BF16(qf.v, b1, s1);
    }
#pragma unroll
    for (int r = 0; r < 8; ++r) {             // D-layout -> LDS strip
      Stile[w][r + 8 * lh][lm]      = s0[r];
      Stile[w][r + 8 * lh][16 + lm] = s1[r];
    }
    // online softmax update; lanes l and l+16 redundantly process row lm
    // (LDS ops are in-order within a wave -> no barrier needed)
    float mx = m_run;
#pragma unroll
    for (int c = 0; c < 32; ++c) {
      float s = scale * Stile[w][lm][c] + brow[kcol0 + c];
      mx = fmaxf(mx, s);
    }
    float alpha = expf(m_run - mx);
    float psum = 0.0f;
#pragma unroll
    for (int c = 0; c < 32; ++c) {
      float s = scale * Stile[w][lm][c] + brow[kcol0 + c];
      float p = expf(s - mx);
      psum += p;
      Abf[w][lm][c] = f2bf(p);
    }
    l_run = l_run * alpha + psum;
    m_run = mx;
    rowscale[w][lm] = alpha;
#pragma unroll
    for (int r = 0; r < 8; ++r) {             // rescale running O accumulators
      float a = rowscale[w][r + 8 * lh];
      accA[r] *= a; accB[r] *= a;
    }
    AFrag af;                                  // probs as A-fragment (16x32)
    af.q[0] = *(const uint4*)&Abf[w][lm][lh * 16];
    af.q[1] = *(const uint4*)&Abf[w][lm][lh * 16 + 8];
    const v16bf v0 = *(const v16bf*)&Vs[lm][kcol0 + lh * 16];
    const v16bf v1 = *(const v16bf*)&Vs[16 + lm][kcol0 + lh * 16];
    accA = WMMA_BF16(af.v, v0, accA);
    accB = WMMA_BF16(af.v, v1, accB);
  }

  rowsum[w][lm] = l_run;
#pragma unroll
  for (int r = 0; r < 8; ++r) {
    int j = j0 + r + 8 * lh;
    float linv = 1.0f / rowsum[w][r + 8 * lh];
    {
      int c = lm;
      float gv = bf2f(gbuf[base + (size_t)c * NPIX + j]);
      ob[(size_t)(c * NHv + h) * NPIX + (size_t)ii * Lv + j] = f2bf(accA[r] * linv * gv);
    }
    {
      int c = 16 + lm;
      float gv = bf2f(gbuf[base + (size_t)c * NPIX + j]);
      ob[(size_t)(c * NHv + h) * NPIX + (size_t)ii * Lv + j] = f2bf(accB[r] * linv * gv);
    }
  }
}

// ---------------------------------------------------------------------------
// Host launch
// ---------------------------------------------------------------------------
extern "C" void kernel_launch(void* const* d_in, const int* in_sizes, int n_in,
                              void* d_out, int out_size, void* d_ws, size_t ws_size,
                              hipStream_t stream) {
  const float* x     = (const float*)d_in[0];
  const float* gamma = (const float*)d_in[1];
  const float* beta  = (const float*)d_in[2];
  const float* w_q   = (const float*)d_in[3];
  const float* w_k   = (const float*)d_in[4];
  const float* w_v   = (const float*)d_in[5];
  const float* w_b   = (const float*)d_in[6];
  const float* w_g   = (const float*)d_in[7];
  const float* b_g   = (const float*)d_in[8];
  const float* w_rep = (const float*)d_in[9];
  const float* b_rep = (const float*)d_in[10];
  float* out = (float*)d_out;
  char* ws = (char*)d_ws;

  // workspace layout (bytes)
  double* part  = (double*)(ws + 0);            //   4 KB
  float*  stats = (float*)(ws + 4096);
  u16*    pw    = (u16*)(ws + 8192);            // 5 * 32 KB packed weights
  float*  biasb = (float*)(ws + 172032);        //   1 MB bias map
  const size_t BIG = 16777216;                  //  16 MB per bf16 plane
  u16* xn   = (u16*)(ws + 1220608);
  u16* qbuf = (u16*)(ws + 1220608 + 1 * BIG);
  u16* kbuf = (u16*)(ws + 1220608 + 2 * BIG);
  u16* vbuf = (u16*)(ws + 1220608 + 3 * BIG);
  u16* gbuf = (u16*)(ws + 1220608 + 4 * BIG);
  u16* obuf = (u16*)(ws + 1220608 + 5 * BIG);

  k_reduce<<<256, 256, 0, stream>>>((const float4*)x, part);
  k_stats<<<1, 256, 0, stream>>>(part, stats);
  k_norm<<<NTOT / 4 / 256, 256, 0, stream>>>((const float4*)x, (const float4*)gamma,
                                             (const float4*)beta, stats, xn);
  k_packw<<<32, 32, 0, stream>>>(w_q,   pw + 0 * 16384);
  k_packw<<<32, 32, 0, stream>>>(w_k,   pw + 1 * 16384);
  k_packw<<<32, 32, 0, stream>>>(w_v,   pw + 2 * 16384);
  k_packw<<<32, 32, 0, stream>>>(w_g,   pw + 3 * 16384);
  k_packw<<<32, 32, 0, stream>>>(w_rep, pw + 4 * 16384);
  k_bias<<<NPIX / 256, 256, 0, stream>>>(xn, w_b, biasb);

  k_gemm<0><<<NPIX / 64, 256, 0, stream>>>(pw + 0 * 16384, xn, (void*)qbuf, nullptr);
  k_gemm<0><<<NPIX / 64, 256, 0, stream>>>(pw + 1 * 16384, xn, (void*)kbuf, nullptr);
  k_gemm<0><<<NPIX / 64, 256, 0, stream>>>(pw + 2 * 16384, xn, (void*)vbuf, nullptr);
  k_gemm<1><<<NPIX / 64, 256, 0, stream>>>(pw + 3 * 16384, xn, (void*)gbuf, b_g);

  k_attn<<<dim3(2, 256, 4), 256, 0, stream>>>(qbuf, kbuf, vbuf, gbuf, biasb, obuf);

  k_gemm<2><<<NPIX / 64, 256, 0, stream>>>(pw + 4 * 16384, obuf, (void*)out, b_rep);
}
